// MicroDVAEModel_19292993093731
// MI455X (gfx1250) — matrix operand order repaired and checked by reading, not compiled, not run on hardware
//
#include <hip/hip_runtime.h>
#include <math.h>

// ---------------- model dims ----------------
#define Bz   4
#define Sz   2048
#define EMB  768
#define Dm   512
#define Hh   8
#define Lz   6
#define DFFz 2048
#define CDz  256
#define NCz  8192
#define Tz   (Bz*Sz)   // 8192 tokens
#define DHz  64

typedef unsigned int  u32;
typedef unsigned short u16;
typedef __bf16 bf16;
typedef __attribute__((ext_vector_type(16))) bf16  v16bf;
typedef __attribute__((ext_vector_type(8)))  float v8f;

union Frag { u32 u[8]; v16bf v; };

__device__ __forceinline__ u16 f2bf(float f) {
  u32 u = __float_as_uint(f);
  u32 r = (u + 0x7FFFu + ((u >> 16) & 1u)) >> 16;   // RNE
  return (u16)r;
}

// A-matrix (16x32 bf16) K position for VGPR v, lane-half (contiguous: k = 16*half + 2*v)
__device__ __forceinline__ int kApos(int v, int half) {
  return half * 16 + 2 * v;
}

__device__ __forceinline__ v8f wmma_bf16(const Frag& a, const Frag& b, v8f c) {
  return __builtin_amdgcn_wmma_f32_16x16x32_bf16(false, a.v, false, b.v,
                                                 (short)0, c, false, false);
}

// CDNA5 async global->LDS copy (bypasses VGPRs, tracked by ASYNCcnt)
__device__ __forceinline__ void async_b128(const u16* gsrc, const u16* ldst) {
  u32 laddr = (u32)(size_t)ldst;     // generic LDS ptr: low 32 bits = DS offset
  asm volatile("global_load_async_to_lds_b128 %0, %1, off"
               :: "v"(laddr), "v"(gsrc) : "memory");
}
__device__ __forceinline__ void wait_async() {
  asm volatile("s_wait_asynccnt 0" ::: "memory");
}

// ---------------- f32 -> bf16 convert ----------------
__global__ void f32_to_bf16_kernel(const float* __restrict__ s,
                                   u16* __restrict__ d, int n) {
  int i = blockIdx.x * 256 + threadIdx.x;
  if (i < n) d[i] = f2bf(s[i]);
}

// ---------------- generic GEMM:  C[M,N] = A[M,K] * W[N,K]^T + bias ----------------
// 128x128 block tile, 32x64 per wave (8 WMMA per K-step), double-buffered
// async-LDS pipeline: one barrier per K-step, DMA of tile i+1 overlaps compute.
__global__ __launch_bounds__(256) void gemm_bf16_kernel(
    const u16* __restrict__ A, const u16* __restrict__ W,
    const float* __restrict__ bias, float* __restrict__ C, u16* __restrict__ Cb,
    int M, int N, int K, int act) {
  __shared__ u16 sA[2][128 * 40];
  __shared__ u16 sW[2][128 * 40];
  const int tid  = threadIdx.x;
  const int wave = tid >> 5, lane = tid & 31;
  const int half = lane >> 4, l15 = lane & 15;
  const int m0 = blockIdx.x * 128, n0 = blockIdx.y * 128;
  const int wm = (wave >> 1) * 32;   // 4 M-groups of 32 rows
  const int wn = (wave & 1) * 64;    // 2 N-groups of 64 cols

  v8f zero = {0.f,0.f,0.f,0.f,0.f,0.f,0.f,0.f};
  v8f acc[2][4];
  #pragma unroll
  for (int mt = 0; mt < 2; ++mt)
    #pragma unroll
    for (int nt = 0; nt < 4; ++nt) acc[mt][nt] = zero;

  auto stage = [&](int buf, int k0) {
    #pragma unroll
    for (int j = 0; j < 2; ++j) {            // 128x32 = 4096 elems per matrix
      int idx = tid * 8 + j * 2048;
      int r = idx >> 5, c = idx & 31;
      async_b128(A + (size_t)(m0 + r) * K + k0 + c, &sA[buf][r * 40 + c]);
      async_b128(W + (size_t)(n0 + r) * K + k0 + c, &sW[buf][r * 40 + c]);
    }
  };

  stage(0, 0);
  int buf = 0;
  for (int k0 = 0; k0 < K; k0 += 32) {
    wait_async();
    __syncthreads();
    if (k0 + 32 < K) stage(buf ^ 1, k0 + 32);

    Frag fa[2], fb[4];
    #pragma unroll
    for (int mt = 0; mt < 2; ++mt) {
      int row = wm + mt * 16 + l15;
      #pragma unroll
      for (int v = 0; v < 8; ++v)
        fa[mt].u[v] = *(const u32*)&sA[buf][row * 40 + kApos(v, half)];
    }
    #pragma unroll
    for (int nt = 0; nt < 4; ++nt) {
      int col = wn + nt * 16 + l15;
      #pragma unroll
      for (int v = 0; v < 8; ++v)
        fb[nt].u[v] = *(const u32*)&sW[buf][col * 40 + half * 16 + 2 * v];
    }
    #pragma unroll
    for (int mt = 0; mt < 2; ++mt)
      #pragma unroll
      for (int nt = 0; nt < 4; ++nt)
        acc[mt][nt] = wmma_bf16(fa[mt], fb[nt], acc[mt][nt]);
    buf ^= 1;
  }

  #pragma unroll
  for (int mt = 0; mt < 2; ++mt)
    #pragma unroll
    for (int nt = 0; nt < 4; ++nt)
      #pragma unroll
      for (int r = 0; r < 8; ++r) {
        int row = m0 + wm + mt * 16 + r + half * 8;
        int col = n0 + wn + nt * 16 + l15;
        float v = acc[mt][nt][r];
        if (bias) v += bias[col];
        if (act)  v = 0.5f * v * (1.0f + erff(v * 0.7071067811865476f));
        size_t o = (size_t)row * N + col;
        if (C)  C[o]  = v;
        if (Cb) Cb[o] = f2bf(v);
      }
}

// ---------------- sinusoid PE add (+bf16 copy) ----------------
__global__ void pe_add_kernel(float* __restrict__ h, u16* __restrict__ hb) {
  int i = blockIdx.x * 256 + threadIdx.x;   // over Tz*Dm
  int d = i & (Dm - 1);
  int t = i >> 9;
  int s = t & (Sz - 1);
  float ang = (float)s * __expf((float)(d & ~1) * (-9.210340372f / 512.0f));
  float pe  = (d & 1) ? cosf(ang) : sinf(ang);
  float y = h[i] + pe;
  h[i]  = y;
  hb[i] = f2bf(y);
}

// ---------------- flash attention ----------------
// Block = 4 waves sharing one (b,h) and 64 queries; K/V blocks (32 keys) staged
// cooperatively into double-buffered LDS via async copies (one barrier / block).
__global__ __launch_bounds__(128) void attn_kernel(const u16* __restrict__ qkvb,
                                                   const int* __restrict__ mask,
                                                   u16* __restrict__ ctxb) {
  __shared__ u16 sK[2][32 * 72];
  __shared__ u16 sV[2][32 * 72];
  __shared__ u16 sP[4][16 * 40];
  const int tid = threadIdx.x;
  const int w = tid >> 5, lane = tid & 31;
  const int half = lane >> 4, l15 = lane & 15;
  const int blk = blockIdx.x;        // B*H*(S/64) = 1024
  const int qg  = blk & 31;          // S/64
  const int bh  = blk >> 5;
  const int b = bh >> 3, h = bh & 7;
  const int tb = b * Sz;
  const int q0 = qg * 64 + w * 16;
  const float scale = 0.125f;        // 1/sqrt(64)

  // Q fragments (16x64 = two 16x32 A-frags), direct global gather
  Frag qf[2];
  #pragma unroll
  for (int c = 0; c < 2; ++c)
    #pragma unroll
    for (int v = 0; v < 8; ++v)
      qf[c].u[v] = *(const u32*)&qkvb[(size_t)(tb + q0 + l15) * 1536 +
                                      h * 64 + c * 32 + kApos(v, half)];

  v8f zero = {0.f,0.f,0.f,0.f,0.f,0.f,0.f,0.f};
  v8f acc[4];
  acc[0] = zero; acc[1] = zero; acc[2] = zero; acc[3] = zero;
  float runmax[8], rsum[8];
  #pragma unroll
  for (int r = 0; r < 8; ++r) { runmax[r] = -1e30f; rsum[r] = 0.f; }

  auto stage = [&](int buf, int kb) {
    #pragma unroll
    for (int j = 0; j < 2; ++j) {            // 32x64 = 2048 elems per matrix
      int idx = tid * 8 + j * 1024;
      int r = idx >> 6, c = idx & 63;
      const u16* g = qkvb + (size_t)(tb + kb + r) * 1536 + h * 64 + c;
      async_b128(g + 512,  &sK[buf][r * 72 + c]);
      async_b128(g + 1024, &sV[buf][r * 72 + c]);
    }
  };

  stage(0, 0);
  int buf = 0;
  for (int kb = 0; kb < Sz; kb += 32) {
    wait_async();
    __syncthreads();
    if (kb + 32 < Sz) stage(buf ^ 1, kb + 32);

    // scores: Q(16x64) x K^T -> two 16x16 tiles (keys kb..+15, +16..+31)
    v8f s0 = zero, s1 = zero;
    #pragma unroll
    for (int c2 = 0; c2 < 2; ++c2) {
      Frag fk0, fk1;
      #pragma unroll
      for (int v = 0; v < 8; ++v) {
        int kdh = c2 * 32 + half * 16 + 2 * v;
        fk0.u[v] = *(const u32*)&sK[buf][ l15       * 72 + kdh];
        fk1.u[v] = *(const u32*)&sK[buf][(16 + l15) * 72 + kdh];
      }
      s0 = wmma_bf16(qf[c2], fk0, s0);
      s1 = wmma_bf16(qf[c2], fk1, s1);
    }
    float bias0 = mask[b * Sz + kb + l15]      ? 0.f : -1e9f;
    float bias1 = mask[b * Sz + kb + 16 + l15] ? 0.f : -1e9f;

    // online softmax over this 32-key block (row stats live in 16-lane halves)
    #pragma unroll
    for (int r = 0; r < 8; ++r) {
      float a0 = s0[r] * scale + bias0;
      float a1 = s1[r] * scale + bias1;
      float mx = fmaxf(a0, a1);
      #pragma unroll
      for (int off = 1; off < 16; off <<= 1) mx = fmaxf(mx, __shfl_xor(mx, off, 32));
      float nm = fmaxf(runmax[r], mx);
      float alpha = __expf(runmax[r] - nm);
      runmax[r] = nm;
      float p0 = __expf(a0 - nm), p1 = __expf(a1 - nm);
      float ls = p0 + p1;
      #pragma unroll
      for (int off = 1; off < 16; off <<= 1) ls += __shfl_xor(ls, off, 32);
      rsum[r] = rsum[r] * alpha + ls;
      #pragma unroll
      for (int ct = 0; ct < 4; ++ct) acc[ct][r] *= alpha;
      sP[w][(r + half * 8) * 40 + l15]      = f2bf(p0);
      sP[w][(r + half * 8) * 40 + 16 + l15] = f2bf(p1);
    }
    // re-fragment P (16x32) as A-matrix via LDS
    Frag pf;
    #pragma unroll
    for (int v = 0; v < 8; ++v)
      pf.u[v] = *(const u32*)&sP[w][l15 * 40 + kApos(v, half)];
    // ctx += P x V (four 16-col tiles of DH)
    #pragma unroll
    for (int ct = 0; ct < 4; ++ct) {
      Frag fv;
      int dhc = ct * 16 + l15;
      #pragma unroll
      for (int v = 0; v < 8; ++v) {
        u32 lo = sV[buf][(half * 16 + 2 * v    ) * 72 + dhc];
        u32 hi = sV[buf][(half * 16 + 2 * v + 1) * 72 + dhc];
        fv.u[v] = lo | (hi << 16);
      }
      acc[ct] = wmma_bf16(pf, fv, acc[ct]);
    }
    buf ^= 1;
  }
  #pragma unroll
  for (int ct = 0; ct < 4; ++ct)
    #pragma unroll
    for (int r = 0; r < 8; ++r) {
      float o = acc[ct][r] / rsum[r];
      ctxb[(size_t)(tb + q0 + r + half * 8) * Dm + h * 64 + ct * 16 + l15] = f2bf(o);
    }
}

// ---------------- fused residual + LayerNorm (one wave per token) ----------------
template <int DN>
__global__ __launch_bounds__(128) void ln_kernel(
    const float* __restrict__ X, const float* __restrict__ R,
    const float* __restrict__ g, const float* __restrict__ bta,
    float* __restrict__ Y, u16* __restrict__ Yb) {
  constexpr int CNT = DN / 32;
  int t = blockIdx.x * 4 + (threadIdx.x >> 5);
  int lane = threadIdx.x & 31;
  const float* xp = X + (size_t)t * DN;
  float xv[CNT];
  float s = 0.f;
  #pragma unroll
  for (int i = 0; i < CNT; ++i) {
    float v = xp[i * 32 + lane];
    if (R) v += R[(size_t)t * DN + i * 32 + lane];
    xv[i] = v; s += v;
  }
  #pragma unroll
  for (int off = 1; off < 32; off <<= 1) s += __shfl_xor(s, off, 32);
  float m = s * (1.0f / DN);
  float vs = 0.f;
  #pragma unroll
  for (int i = 0; i < CNT; ++i) { float d = xv[i] - m; vs += d * d; }
  #pragma unroll
  for (int off = 1; off < 32; off <<= 1) vs += __shfl_xor(vs, off, 32);
  float rstd = rsqrtf(vs * (1.0f / DN) + 1e-5f);
  #pragma unroll
  for (int i = 0; i < CNT; ++i) {
    int d = i * 32 + lane;
    float y = (xv[i] - m) * rstd * g[d] + bta[d];
    if (Y)  Y[(size_t)t * DN + d]  = y;
    if (Yb) Yb[(size_t)t * DN + d] = f2bf(y);
  }
}

// ---------------- codebook row-normalize -> bf16 ----------------
__global__ __launch_bounds__(128) void cb_norm_kernel(const float* __restrict__ cb,
                                                      u16* __restrict__ cnb) {
  int row  = blockIdx.x * 4 + (threadIdx.x >> 5);
  int lane = threadIdx.x & 31;
  float vals[8]; float ss = 0.f;
  #pragma unroll
  for (int i = 0; i < 8; ++i) {
    float v = cb[(size_t)row * CDz + i * 32 + lane];
    vals[i] = v; ss += v * v;
  }
  #pragma unroll
  for (int off = 1; off < 32; off <<= 1) ss += __shfl_xor(ss, off, 32);
  float inv = 1.f / fmaxf(sqrtf(ss), 1e-8f);
  #pragma unroll
  for (int i = 0; i < 8; ++i)
    cnb[(size_t)row * CDz + i * 32 + lane] = f2bf(vals[i] * inv);
}

// ---------------- VQ argmax ----------------
// 4 waves (4 token tiles) share async-staged codebook tiles; online argmax.
__global__ __launch_bounds__(128) void vq_argmax_kernel(const u16* __restrict__ zb,
                                                        const u16* __restrict__ cnb,
                                                        const int* __restrict__ mask,
                                                        int* __restrict__ idx_out) {
  __shared__ u16 sCB[2][16 * 264];
  const int tid = threadIdx.x;
  const int w = tid >> 5, lane = tid & 31;
  const int half = lane >> 4, l15 = lane & 15;
  const int g = blockIdx.x * 4 + w;   // 0..511 token tiles
  const int t0 = g * 16;

  Frag za[8];   // resident A-frags for K=256
  #pragma unroll
  for (int c = 0; c < 8; ++c)
    #pragma unroll
    for (int v = 0; v < 8; ++v)
      za[c].u[v] = *(const u32*)&zb[(size_t)(t0 + l15) * CDz + c * 32 + kApos(v, half)];

  float bv[8]; int bi[8];
  #pragma unroll
  for (int r = 0; r < 8; ++r) { bv[r] = -1e30f; bi[r] = 0; }
  v8f zero = {0.f,0.f,0.f,0.f,0.f,0.f,0.f,0.f};

  auto stage = [&](int buf, int ct) {
    #pragma unroll
    for (int j = 0; j < 4; ++j) {            // 16x256 = 4096 elems
      int idx = tid * 8 + j * 1024;
      int r = idx >> 8, c = idx & 255;
      async_b128(cnb + (size_t)(ct * 16 + r) * CDz + c, &sCB[buf][r * 264 + c]);
    }
  };

  stage(0, 0);
  int buf = 0;
  for (int ct = 0; ct < NCz / 16; ++ct) {
    wait_async();
    __syncthreads();
    if (ct + 1 < NCz / 16) stage(buf ^ 1, ct + 1);

    v8f cacc = zero;
    #pragma unroll
    for (int c = 0; c < 8; ++c) {
      Frag fb;
      #pragma unroll
      for (int v = 0; v < 8; ++v)
        fb.u[v] = *(const u32*)&sCB[buf][l15 * 264 + c * 32 + half * 16 + 2 * v];
      cacc = wmma_bf16(za[c], fb, cacc);
    }
    int code = ct * 16 + l15;
    #pragma unroll
    for (int r = 0; r < 8; ++r)
      if (cacc[r] > bv[r]) { bv[r] = cacc[r]; bi[r] = code; }
    buf ^= 1;
  }
  // cross-lane argmax within each 16-lane half; first-index tie-break
  #pragma unroll
  for (int r = 0; r < 8; ++r)
    #pragma unroll
    for (int off = 1; off < 16; off <<= 1) {
      float ov = __shfl_xor(bv[r], off, 32);
      int   oi = __shfl_xor(bi[r], off, 32);
      if (ov > bv[r] || (ov == bv[r] && oi < bi[r])) { bv[r] = ov; bi[r] = oi; }
    }
  if (l15 == 0) {
    #pragma unroll
    for (int r = 0; r < 8; ++r) {
      int t = t0 + r + half * 8;
      idx_out[t] = (mask[t] == 0) ? 0 : bi[r];
    }
  }
}

// ---------------- host orchestration ----------------
extern "C" void kernel_launch(void* const* d_in, const int* in_sizes, int n_in,
                              void* d_out, int out_size, void* d_ws, size_t ws_size,
                              hipStream_t stream) {
  (void)in_sizes; (void)n_in; (void)out_size; (void)ws_size;
  const float* x      = (const float*)d_in[0];
  const int*   mask   = (const int*)  d_in[1];
  const float* W_in   = (const float*)d_in[2];
  const float* b_in   = (const float*)d_in[3];
  const float* Wqkv   = (const float*)d_in[4];
  const float* bqkv   = (const float*)d_in[5];
  const float* Wo     = (const float*)d_in[6];
  const float* bo     = (const float*)d_in[7];
  const float* g1     = (const float*)d_in[8];
  const float* bln1   = (const float*)d_in[9];
  const float* W1     = (const float*)d_in[10];
  const float* b1     = (const float*)d_in[11];
  const float* W2     = (const float*)d_in[12];
  const float* b2     = (const float*)d_in[13];
  const float* g2     = (const float*)d_in[14];
  const float* bln2   = (const float*)d_in[15];
  const float* g_enc  = (const float*)d_in[16];
  const float* b_enc  = (const float*)d_in[17];
  const float* W_pv   = (const float*)d_in[18];
  const float* b_pv   = (const float*)d_in[19];
  const float* g_pv   = (const float*)d_in[20];
  const float* bln_pv = (const float*)d_in[21];
  const float* cbk    = (const float*)d_in[22];

  int*   idx_out = (int*)d_out;
  float* z_out   = (float*)d_out + Tz;   // z follows indices (both 4B elems)

  char* base = (char*)d_ws; size_t off = 0;
  auto alloc = [&](size_t bytes) -> void* {
    void* p = base + off; off = (off + bytes + 255) & ~(size_t)255; return p;
  };
  u16*   xb    = (u16*)  alloc((size_t)Tz * EMB * 2);
  u16*   W_inb = (u16*)  alloc((size_t)Dm * EMB * 2);
  u16*   Wqkvb = (u16*)  alloc((size_t)Lz * 3 * Dm * Dm * 2);
  u16*   Wob   = (u16*)  alloc((size_t)Lz * Dm * Dm * 2);
  u16*   W1b   = (u16*)  alloc((size_t)Lz * DFFz * Dm * 2);
  u16*   W2b   = (u16*)  alloc((size_t)Lz * Dm * DFFz * 2);
  u16*   W_pvb = (u16*)  alloc((size_t)CDz * Dm * 2);
  u16*   cnb   = (u16*)  alloc((size_t)NCz * CDz * 2);
  float* h     = (float*)alloc((size_t)Tz * Dm * 4);
  u16*   hb    = (u16*)  alloc((size_t)Tz * Dm * 2);
  u16*   qkvb  = (u16*)  alloc((size_t)Tz * 3 * Dm * 2);
  u16*   ctxb  = (u16*)  alloc((size_t)Tz * Dm * 2);
  float* tmp   = (float*)alloc((size_t)Tz * Dm * 4);
  u16*   ub    = (u16*)  alloc((size_t)Tz * DFFz * 2);
  float* zpre  = (float*)alloc((size_t)Tz * CDz * 4);
  u16*   zb    = (u16*)  alloc((size_t)Tz * CDz * 2);

  auto cvt = [&](const float* s, u16* d, size_t n) {
    f32_to_bf16_kernel<<<(unsigned)((n + 255) / 256), 256, 0, stream>>>(s, d, (int)n);
  };
  cvt(x,    xb,    (size_t)Tz * EMB);
  cvt(W_in, W_inb, (size_t)Dm * EMB);
  cvt(Wqkv, Wqkvb, (size_t)Lz * 3 * Dm * Dm);
  cvt(Wo,   Wob,   (size_t)Lz * Dm * Dm);
  cvt(W1,   W1b,   (size_t)Lz * DFFz * Dm);
  cvt(W2,   W2b,   (size_t)Lz * Dm * DFFz);
  cvt(W_pv, W_pvb, (size_t)CDz * Dm);
  cb_norm_kernel<<<NCz / 4, 128, 0, stream>>>(cbk, cnb);

  auto gemm = [&](const u16* A, const u16* Wt, const float* bias, float* C, u16* Cb,
                  int M, int N, int K, int act) {
    dim3 grid(M / 128, N / 128);
    gemm_bf16_kernel<<<grid, 256, 0, stream>>>(A, Wt, bias, C, Cb, M, N, K, act);
  };

  // input projection + positional encoding
  gemm(xb, W_inb, b_in, h, nullptr, Tz, Dm, EMB, 0);
  pe_add_kernel<<<Tz * Dm / 256, 256, 0, stream>>>(h, hb);

  for (int l = 0; l < Lz; ++l) {
    gemm(hb, Wqkvb + (size_t)l * 3 * Dm * Dm, bqkv + (size_t)l * 3 * Dm,
         nullptr, qkvb, Tz, 3 * Dm, Dm, 0);
    attn_kernel<<<Bz * Hh * (Sz / 64), 128, 0, stream>>>(qkvb, mask, ctxb);
    gemm(ctxb, Wob + (size_t)l * Dm * Dm, bo + (size_t)l * Dm,
         tmp, nullptr, Tz, Dm, Dm, 0);
    ln_kernel<512><<<Tz / 4, 128, 0, stream>>>(tmp, h, g1 + (size_t)l * Dm,
                                               bln1 + (size_t)l * Dm, h, hb);
    gemm(hb, W1b + (size_t)l * DFFz * Dm, b1 + (size_t)l * DFFz,
         nullptr, ub, Tz, DFFz, Dm, 1);
    gemm(ub, W2b + (size_t)l * Dm * DFFz, b2 + (size_t)l * Dm,
         tmp, nullptr, Tz, Dm, DFFz, 0);
    ln_kernel<512><<<Tz / 4, 128, 0, stream>>>(tmp, h, g2 + (size_t)l * Dm,
                                               bln2 + (size_t)l * Dm, h, hb);
  }
  ln_kernel<512><<<Tz / 4, 128, 0, stream>>>(h, nullptr, g_enc, b_enc, nullptr, hb);
  gemm(hb, W_pvb, b_pv, zpre, nullptr, Tz, CDz, Dm, 0);
  ln_kernel<256><<<Tz / 4, 128, 0, stream>>>(zpre, nullptr, g_pv, bln_pv, z_out, zb);
  vq_argmax_kernel<<<(Tz / 16) / 4, 128, 0, stream>>>(zb, cnb, mask, idx_out);
}
